// SlidingWindowTransformerLayer_57956288692767
// MI455X (gfx1250) — matrix operand order repaired and checked by reading, not compiled, not run on hardware
//
#include <hip/hip_runtime.h>
#include <hip/hip_bf16.h>
#include <cstdint>

// Problem constants (reference: B=4, T=2048, D=512, H=8, F=2048, WINDOW=128)
#define B_   4
#define T_   2048
#define D_   512
#define H_   8
#define F_   2048
#define HD_  64
#define WIN_ 128
#define BT_  (B_ * T_)

typedef __attribute__((ext_vector_type(16))) _Float16 v16h;
typedef __attribute__((ext_vector_type(8)))  _Float16 v8h;
typedef __attribute__((ext_vector_type(8)))  float    v8f;
typedef __attribute__((ext_vector_type(4)))  int      v4i;

// ---------------------------------------------------------------------------
// CDNA5 async memory->LDS path (ASYNCcnt-tracked), with sync fallback.
// clang-22 signature: (int4 AS1* src, int4 AS3* dst, imm offset, imm cpol)
// ---------------------------------------------------------------------------
#if __has_builtin(__builtin_amdgcn_global_load_async_to_lds_b128)
#define HAVE_ASYNC 1
#else
#define HAVE_ASYNC 0
#endif

#if HAVE_ASYNC
typedef v4i __attribute__((address_space(1))) as1_v4i;
typedef v4i __attribute__((address_space(3))) as3_v4i;

__device__ __forceinline__ void async_cp16(const _Float16* g, _Float16* l) {
  // AS1 pointer is 64-bit; AS3 pointer is 32-bit (flat LDS addr truncates, ISA 10.2)
  __builtin_amdgcn_global_load_async_to_lds_b128(
      (as1_v4i*)(uintptr_t)g,
      (as3_v4i*)(uint32_t)(uintptr_t)l,
      0, 0);
}
#if __has_builtin(__builtin_amdgcn_s_wait_asynccnt)
#define WAIT_ASYNC(n) __builtin_amdgcn_s_wait_asynccnt(n)
#else
#define WAIT_ASYNC(n) asm volatile("s_wait_asynccnt %0" ::"i"(n) : "memory")
#endif
#endif

// ---------------------------------------------------------------------------
// WMMA fragment helpers (CDNA5 16-bit layouts, ISA 7.12.2)
// A (16x32 f16): lane m = lane&15; lanes>=16 take K+8. Pair p:
//   p 0..3 -> K = 2*p + w ; p 4..7 -> K = 16 + 2*(p-4) + w  (+8 if hi lane)
// B (32x16 f16): lane&15 = N; element i -> K as for A (lanes>=16 take K+8).
// C/D (16x16 f32): element r -> row r + 8*(lane>=16), col = lane&15.
// ---------------------------------------------------------------------------
__device__ __forceinline__ v16h load_a_frag(const _Float16* __restrict__ rowp, int hi) {
  v16h a;
#pragma unroll
  for (int p = 0; p < 8; ++p) {
    int k = ((p & 3) << 1) + ((p & 4) ? 16 : 0) + (hi ? 8 : 0);
    a[2 * p]     = rowp[k];
    a[2 * p + 1] = rowp[k + 1];
  }
  return a;
}

__device__ __forceinline__ v16h load_16h(const _Float16* __restrict__ p) {
  v16h b;
#pragma unroll
  for (int i = 0; i < 16; ++i) b[i] = p[i];
  return b;
}

// B fragment from a row-major [N][K] LDS tile: lane supplies N (lane&15), elements walk K (table order).
__device__ __forceinline__ v16h load_b_frag_rm(const _Float16* __restrict__ base,
                                               int n0, int ldk) {
  v16h b;
  const int n = n0 + (threadIdx.x & 15);
  const int hk = (threadIdx.x & 16) ? 8 : 0;
#pragma unroll
  for (int i = 0; i < 16; ++i) b[i] = base[n * ldk + (i & 7) + ((i & 8) ? 16 : 0) + hk];
  return b;
}

// ---------------------------------------------------------------------------
// f32 -> f16 conversion (weights)
// ---------------------------------------------------------------------------
__global__ void f32_to_f16_kernel(const float* __restrict__ src,
                                  _Float16* __restrict__ dst, int n) {
  int i = blockIdx.x * blockDim.x + threadIdx.x;
  if (i < n) dst[i] = (_Float16)src[i];
}

// ---------------------------------------------------------------------------
// LayerNorm over D=512, one row per block (256 threads = 8 waves), f16 out
// ---------------------------------------------------------------------------
__global__ __launch_bounds__(256) void layernorm_f16_kernel(
    const float* __restrict__ x, const float* __restrict__ g,
    const float* __restrict__ bb, _Float16* __restrict__ out) {
  __shared__ float ssum[8], ssq[8];
  const int row = blockIdx.x;
  const int tid = threadIdx.x;
  const float* xr = x + (size_t)row * D_;
  float v0 = xr[tid], v1 = xr[tid + 256];
  float s = v0 + v1, s2 = v0 * v0 + v1 * v1;
#pragma unroll
  for (int msk = 16; msk >= 1; msk >>= 1) {
    s  += __shfl_xor(s,  msk, 32);
    s2 += __shfl_xor(s2, msk, 32);
  }
  const int lane = tid & 31, wid = tid >> 5;
  if (lane == 0) { ssum[wid] = s; ssq[wid] = s2; }
  __syncthreads();
  float ts = 0.f, tq = 0.f;
#pragma unroll
  for (int w = 0; w < 8; ++w) { ts += ssum[w]; tq += ssq[w]; }
  float mu   = ts * (1.0f / (float)D_);
  float var  = tq * (1.0f / (float)D_) - mu * mu;
  float rstd = rsqrtf(var + 1e-5f);
  out[(size_t)row * D_ + tid]       = (_Float16)((v0 - mu) * rstd * g[tid] + bb[tid]);
  out[(size_t)row * D_ + tid + 256] = (_Float16)((v1 - mu) * rstd * g[tid + 256] + bb[tid + 256]);
}

// ---------------------------------------------------------------------------
// Tiled WMMA GEMM: C[M,N] = A[M,K] * W[N,K]^T + bias (+gelu)(+residual)
// 256 threads = 8 waves; block tile 128x64; K step 32; double-buffered LDS
// fed by GLOBAL_LOAD_ASYNC_TO_LDS_B128 (ASYNCcnt counted-wait pipelining).
// ---------------------------------------------------------------------------
#define BM 128
#define BN 64
#define BK 32

__global__ __launch_bounds__(256) void gemm_wmma_kernel(
    const _Float16* __restrict__ A,  // [M,K]
    const _Float16* __restrict__ Wt, // [N,K] row-major (i.e. W, used as B^T)
    const float* __restrict__ bias,  // [N]
    const float* __restrict__ resid, // [M,N] f32 or nullptr
    float* __restrict__ outF,        // [M,N] f32 or nullptr
    _Float16* __restrict__ outH,     // [M,N] f16 or nullptr
    int M, int N, int K, int act_gelu) {
  __shared__ _Float16 sA[2][BM * BK];  // [row][k] row-major
  __shared__ _Float16 sB[2][BN * BK];  // [n][k]  row-major

  const int tid  = threadIdx.x;
  const int lane = tid & 31, wid = tid >> 5;
  const int wm = wid & 3, wn = wid >> 2;
  const int bm = blockIdx.y * BM, bn = blockIdx.x * BN;
  const int m = lane & 15, hi = lane >> 4;
  const int ar = tid >> 2;          // 0..63
  const int ac = (tid & 3) << 3;    // 0,8,16,24

  v8f acc[2][2];
#pragma unroll
  for (int i = 0; i < 2; ++i)
#pragma unroll
    for (int j = 0; j < 2; ++j)
#pragma unroll
      for (int r = 0; r < 8; ++r) acc[i][j][r] = 0.f;

  const _Float16* gA0 = A  + (size_t)(bm + ar) * K + ac;
  const _Float16* gA1 = A  + (size_t)(bm + ar + 64) * K + ac;
  const _Float16* gB  = Wt + (size_t)(bn + ar) * K + ac;

  // Stage one 128x32 A tile + 64x32 B tile: 3 async (or sync) 16B copies/thread.
  auto stage = [&](int buf, int k0) {
#if HAVE_ASYNC
    async_cp16(gA0 + k0, &sA[buf][ar * BK + ac]);
    async_cp16(gA1 + k0, &sA[buf][(ar + 64) * BK + ac]);
    async_cp16(gB  + k0, &sB[buf][ar * BK + ac]);
#else
    v8h a0 = *(const v8h*)(gA0 + k0);
    v8h a1 = *(const v8h*)(gA1 + k0);
    v8h b0 = *(const v8h*)(gB + k0);
    *(v8h*)&sA[buf][ar * BK + ac]        = a0;
    *(v8h*)&sA[buf][(ar + 64) * BK + ac] = a1;
    *(v8h*)&sB[buf][ar * BK + ac]        = b0;
#endif
  };

  stage(0, 0);
  int cur = 0;
  for (int k0 = 0; k0 < K; k0 += BK) {
    const bool has_next = (k0 + BK) < K;
    if (has_next) {
      stage(cur ^ 1, k0 + BK);  // prefetch next tile while current computes
#if !HAVE_ASYNC
      __builtin_prefetch(gA0 + k0 + 2 * BK, 0, 1);
#endif
    }
#if HAVE_ASYNC
    if (has_next) WAIT_ASYNC(3);   // leave next tile's 3 copies in flight
    else          WAIT_ASYNC(0);
#endif
    __syncthreads();

    v16h af[2], bf[2];
#pragma unroll
    for (int i = 0; i < 2; ++i)
      af[i] = load_a_frag(&sA[cur][(wm * 32 + i * 16 + m) * BK], hi);
#pragma unroll
    for (int j = 0; j < 2; ++j)
      bf[j] = load_b_frag_rm(&sB[cur][0], wn * 32 + j * 16, BK);

#pragma unroll
    for (int i = 0; i < 2; ++i)
#pragma unroll
      for (int j = 0; j < 2; ++j)
        acc[i][j] = __builtin_amdgcn_wmma_f32_16x16x32_f16(
            false, af[i], false, bf[j], (short)0, acc[i][j], false, false);
    __syncthreads();  // all waves done reading before buffer is recycled
    cur ^= 1;
  }

  // Epilogue: bias (+gelu)(+residual), write f32 and/or f16
#pragma unroll
  for (int i = 0; i < 2; ++i)
#pragma unroll
    for (int j = 0; j < 2; ++j)
#pragma unroll
      for (int r = 0; r < 8; ++r) {
        int row = bm + wm * 32 + i * 16 + r + hi * 8;
        int col = bn + wn * 32 + j * 16 + m;
        float v = acc[i][j][r] + bias[col];
        if (act_gelu) v = 0.5f * v * (1.0f + erff(v * 0.70710678118654752f));
        size_t o = (size_t)row * N + col;
        if (resid) v += resid[o];
        if (outF) outF[o] = v;
        if (outH) outH[o] = (_Float16)v;
      }
}

// ---------------------------------------------------------------------------
// Fused sliding-window attention (flash-style, online softmax).
// One wave (32 threads) per (b, h, 16-query tile). Key chunks of 32.
// qkv: [B, T, 3*D] f16 (q | k | v, each head HD=64).  ctx: [B, T, D] f16.
// K/V chunks staged to LDS via async copies (row-major).
// ---------------------------------------------------------------------------
__global__ __launch_bounds__(32) void attn_swa_kernel(
    const _Float16* __restrict__ qkv, _Float16* __restrict__ ctx) {
  __shared__ _Float16 sK[32 * HD_];   // [key][dim]
  __shared__ _Float16 sV[32 * HD_];   // [key][dim]
  __shared__ _Float16 sP[16 * 32];    // probs (for A-frag transpose)

  const int lane = threadIdx.x;
  const int m = lane & 15, hi = lane >> 4;
  const int q0 = blockIdx.x * 16;
  const int b  = blockIdx.y >> 3;
  const int h  = blockIdx.y & 7;

  const _Float16* base = qkv + (size_t)b * T_ * (3 * D_);
  const _Float16* qrow = base + (size_t)(q0 + m) * (3 * D_) + h * HD_;

  v16h qf[2];
#pragma unroll
  for (int p = 0; p < 2; ++p) qf[p] = load_a_frag(qrow + p * 32, hi);

  v8f acc[4];
#pragma unroll
  for (int n = 0; n < 4; ++n)
#pragma unroll
    for (int r = 0; r < 8; ++r) acc[n][r] = 0.f;

  float mx[8], lsum[8];
#pragma unroll
  for (int r = 0; r < 8; ++r) { mx[r] = -3.0e38f; lsum[r] = 0.f; }

  int kstart = q0 - (WIN_ - 1);
  if (kstart < 0) kstart = 0;
  kstart &= ~31;

  for (int kb = kstart; kb < q0 + 16; kb += 32) {
    // Stage 32 K rows + 32 V rows (clamped reads; mask handles the tail)
    int key = kb + lane;
    if (key > T_ - 1) key = T_ - 1;
    const _Float16* krow = base + (size_t)key * (3 * D_) + D_     + h * HD_;
    const _Float16* vrow = base + (size_t)key * (3 * D_) + 2 * D_ + h * HD_;
#if HAVE_ASYNC
#pragma unroll
    for (int t = 0; t < 8; ++t) {
      async_cp16(krow + t * 8, &sK[lane * HD_ + t * 8]);
      async_cp16(vrow + t * 8, &sV[lane * HD_ + t * 8]);
    }
    WAIT_ASYNC(0);
#else
#pragma unroll
    for (int t = 0; t < 8; ++t) {
      *(v8h*)(sK + lane * HD_ + t * 8) = *(const v8h*)(krow + t * 8);
      *(v8h*)(sV + lane * HD_ + t * 8) = *(const v8h*)(vrow + t * 8);
    }
#endif
    __syncthreads();

    // S(16x32) = Q(16x64) x K^T(64x32): 2 N-tiles x 2 K-pieces = 4 WMMAs
    v8f s[2];
#pragma unroll
    for (int n = 0; n < 2; ++n) {
#pragma unroll
      for (int r = 0; r < 8; ++r) s[n][r] = 0.f;
#pragma unroll
      for (int p = 0; p < 2; ++p) {
        v16h bf = load_b_frag_rm(sK + p * 32, n * 16, HD_);
        s[n] = __builtin_amdgcn_wmma_f32_16x16x32_f16(
            false, qf[p], false, bf, (short)0, s[n], false, false);
      }
    }

    // Online softmax (rows split across lane halves; xor masks 1..8 stay in-half)
    const float scale = 0.125f;  // 1/sqrt(64)
    float pe[2][8];
#pragma unroll
    for (int r = 0; r < 8; ++r) {
      int ig = q0 + r + hi * 8;
      float rm = -3.0e38f;
#pragma unroll
      for (int n = 0; n < 2; ++n) {
        int jg = kb + n * 16 + m;
        bool ok = (jg <= ig) && (jg >= ig - (WIN_ - 1));
        float v = ok ? s[n][r] * scale : -3.0e38f;
        pe[n][r] = v;
        rm = fmaxf(rm, v);
      }
#pragma unroll
      for (int msk = 8; msk >= 1; msk >>= 1) rm = fmaxf(rm, __shfl_xor(rm, msk, 32));
      float mn = fmaxf(mx[r], rm);
      float sf = __expf(mx[r] - mn);
      float rs = 0.f;
#pragma unroll
      for (int n = 0; n < 2; ++n) {
        int jg = kb + n * 16 + m;
        bool ok = (jg <= ig) && (jg >= ig - (WIN_ - 1));
        float e = ok ? __expf(pe[n][r] - mn) : 0.f;
        pe[n][r] = e;
        rs += e;
      }
#pragma unroll
      for (int msk = 8; msk >= 1; msk >>= 1) rs += __shfl_xor(rs, msk, 32);
      lsum[r] = lsum[r] * sf + rs;
      mx[r] = mn;
#pragma unroll
      for (int n = 0; n < 4; ++n) acc[n][r] *= sf;
    }

    // Transpose P through LDS into A-fragment layout
#pragma unroll
    for (int r = 0; r < 8; ++r)
#pragma unroll
      for (int n = 0; n < 2; ++n)
        sP[(r + hi * 8) * 32 + n * 16 + m] = (_Float16)pe[n][r];
    __syncthreads();

    v16h pf = load_a_frag(&sP[m * 32], hi);
#pragma unroll
    for (int n = 0; n < 4; ++n) {
      v16h vf;
#pragma unroll
      for (int i = 0; i < 16; ++i)
        vf[i] = sV[((i & 7) + ((i & 8) ? 16 : 0) + hi * 8) * HD_ + n * 16 + m];
      acc[n] = __builtin_amdgcn_wmma_f32_16x16x32_f16(
          false, pf, false, vf, (short)0, acc[n], false, false);
    }
    __syncthreads();
  }

  // Normalize and store context (f16)
#pragma unroll
  for (int n = 0; n < 4; ++n)
#pragma unroll
    for (int r = 0; r < 8; ++r) {
      int row = q0 + r + hi * 8;
      float o = acc[n][r] / lsum[r];
      ctx[((size_t)b * T_ + row) * D_ + h * HD_ + n * 16 + m] = (_Float16)o;
    }
}

// ---------------------------------------------------------------------------
// Launch
// ---------------------------------------------------------------------------
extern "C" void kernel_launch(void* const* d_in, const int* in_sizes, int n_in,
                              void* d_out, int out_size, void* d_ws, size_t ws_size,
                              hipStream_t stream) {
  (void)in_sizes; (void)n_in; (void)out_size; (void)ws_size;
  const float* x         = (const float*)d_in[0];
  const float* in_proj_w = (const float*)d_in[1];
  const float* in_proj_b = (const float*)d_in[2];
  const float* out_w     = (const float*)d_in[3];
  const float* out_b     = (const float*)d_in[4];
  const float* ln1_g     = (const float*)d_in[5];
  const float* ln1_b     = (const float*)d_in[6];
  const float* ln2_g     = (const float*)d_in[7];
  const float* ln2_b     = (const float*)d_in[8];
  const float* w1        = (const float*)d_in[9];
  const float* b1        = (const float*)d_in[10];
  const float* w2        = (const float*)d_in[11];
  const float* b2        = (const float*)d_in[12];

  char* p = (char*)d_ws;
  auto alloc = [&](size_t bytes) {
    char* r = p;
    p += (bytes + 255) & ~(size_t)255;
    return r;
  };
  _Float16* x2h  = (_Float16*)alloc((size_t)BT_ * D_ * 2);
  _Float16* wqkv = (_Float16*)alloc((size_t)3 * D_ * D_ * 2);
  _Float16* qkvh = (_Float16*)alloc((size_t)BT_ * 3 * D_ * 2);
  _Float16* ctxh = (_Float16*)alloc((size_t)BT_ * D_ * 2);
  _Float16* woh  = (_Float16*)alloc((size_t)D_ * D_ * 2);
  float*    xres = (float*)   alloc((size_t)BT_ * D_ * 4);
  _Float16* x3h  = (_Float16*)alloc((size_t)BT_ * D_ * 2);
  _Float16* w1h  = (_Float16*)alloc((size_t)F_ * D_ * 2);
  _Float16* hh   = (_Float16*)alloc((size_t)BT_ * F_ * 2);
  _Float16* w2h  = (_Float16*)alloc((size_t)D_ * F_ * 2);

  auto conv = [&](const float* s, _Float16* d, int n) {
    f32_to_f16_kernel<<<(n + 255) / 256, 256, 0, stream>>>(s, d, n);
  };
  conv(in_proj_w, wqkv, 3 * D_ * D_);
  conv(out_w, woh, D_ * D_);
  conv(w1, w1h, F_ * D_);
  conv(w2, w2h, D_ * F_);

  // 1) LN1 -> x2 (f16)
  layernorm_f16_kernel<<<BT_, 256, 0, stream>>>(x, ln1_g, ln1_b, x2h);

  // 2) QKV = x2 @ in_proj_w^T + b  -> f16 [BT, 1536]
  gemm_wmma_kernel<<<dim3((3 * D_) / BN, BT_ / BM), 256, 0, stream>>>(
      x2h, wqkv, in_proj_b, nullptr, nullptr, qkvh, BT_, 3 * D_, D_, 0);

  // 3) Sliding-window attention -> ctx f16 [BT, 512]
  attn_swa_kernel<<<dim3(T_ / 16, B_ * H_), 32, 0, stream>>>(qkvh, ctxh);

  // 4) x_res = x + ctx @ out_w^T + out_b   (f32)
  gemm_wmma_kernel<<<dim3(D_ / BN, BT_ / BM), 256, 0, stream>>>(
      ctxh, woh, out_b, x, xres, nullptr, BT_, D_, D_, 0);

  // 5) LN2 -> x3 (f16)
  layernorm_f16_kernel<<<BT_, 256, 0, stream>>>(xres, ln2_g, ln2_b, x3h);

  // 6) h = gelu(x3 @ w1^T + b1) -> f16 [BT, 2048]
  gemm_wmma_kernel<<<dim3(F_ / BN, BT_ / BM), 256, 0, stream>>>(
      x3h, w1h, b1, nullptr, nullptr, hh, BT_, F_, D_, 1);

  // 7) out = x_res + h @ w2^T + b2 -> f32 d_out
  gemm_wmma_kernel<<<dim3(D_ / BN, BT_ / BM), 256, 0, stream>>>(
      hh, w2h, b2, xres, (float*)d_out, nullptr, BT_, D_, F_, 0);
}
